// BasicModel_29652454212536
// MI455X (gfx1250) — compile-verified, matrix operand order checked by reading, not run on hardware
//
#include <hip/hip_runtime.h>
#include <math.h>

typedef __attribute__((ext_vector_type(16))) _Float16 v16h;
typedef __attribute__((ext_vector_type(8)))  float    v8f;

#define NP   100000      // particles (GEMM M)
#define KC   64          // communities (GEMM N)
#define OD   1024        // OTUs (GEMM K / reduction dim)
#define EPSF 1e-6f

#define ROWS_PER_WAVE   32                                     // two 16-row M tiles
#define WAVES_PER_BLOCK 8
#define ROWS_PER_BLOCK  (ROWS_PER_WAVE * WAVES_PER_BLOCK)      // 256
#define NBLOCKS         ((NP + ROWS_PER_BLOCK - 1) / ROWS_PER_BLOCK) // 391
#define KT_COUNT        (OD / 32)                              // 32 k-steps
#define CT_COUNT        (KC / 16)                              // 4 column tiles

// Packed-B: fragment (kt,ct) -> 32 lanes x 16 halves (v16h per lane)
#define BPACK_HALFS  (KT_COUNT * CT_COUNT * 32 * 16)           // 65536 halves
#define WS_BPACK_BYTES (BPACK_HALFS * 2)                       // 131072 B

// ---------------------------------------------------------------------------
// Kernel 1: B = log(otu_dist + eps) in f16, pre-swizzled to the WMMA B-operand
// per-lane layout (mirror of ISA 16-bit A 16x32 layout, M->N).
// Element index t encodes: t = ((kt*4+ct)*32 + lane)*16 + e  (contiguous store)
// ---------------------------------------------------------------------------
__global__ __launch_bounds__(256) void pack_logB(const float* __restrict__ otu,
                                                 _Float16* __restrict__ bpack) {
    int t    = blockIdx.x * blockDim.x + threadIdx.x;   // 0 .. 65535
    int e    = t & 15;            // element within lane's v16h
    int l    = (t >> 4) & 31;     // lane
    int frag = t >> 9;            // fragment id
    int ct   = frag & (CT_COUNT - 1);
    int kt   = frag / CT_COUNT;

    int half = l >> 4;
    int nl   = l & 15;
    int c    = ct * 16 + nl;      // community (GEMM N index)
    int v    = e >> 1;            // VGPR within operand
    int w    = e & 1;             // half within dword
    int klocal = (v < 4 ? 2 * v : 16 + 2 * (v - 4)) + w + 8 * half;
    int o    = kt * 32 + klocal;  // OTU (GEMM K index)

    bpack[t] = (_Float16)__logf(otu[(size_t)c * OD + o] + EPSF);
}

// ---------------------------------------------------------------------------
// Kernel 2: fused GEMM (WMMA f16 -> f32) + log-prior + logsumexp + block sum.
// Each wave: 32 rows (two M tiles sharing B fragments), 8 WMMAs per k-step.
// Dynamic LDS layout:
//   [0, 131072)          : staged packed B  (reused as 64KB row buffer later)
//   [131072, 131072+256) : log prior (64 floats)
//   [131328, 131360)     : per-wave partials (8 floats)
// ---------------------------------------------------------------------------
__global__ __launch_bounds__(256) void ll_fused(const float* __restrict__ counts,
                                                const float* __restrict__ comm,
                                                const _Float16* __restrict__ bpack,
                                                float* __restrict__ partials) {
    extern __shared__ char smem[];
    _Float16* ldsB     = (_Float16*)smem;
    float*    ldsRows  = (float*)smem;                 // reused after barrier
    float*    ldsPrior = (float*)(smem + WS_BPACK_BYTES);
    float*    ldsWave  = (float*)(smem + WS_BPACK_BYTES + 256);

    const int tid  = threadIdx.x;
    const int wave = tid >> 5;
    const int lane = tid & 31;
    const int half = lane >> 4;
    const int nl   = lane & 15;

    // ---- stage packed B into LDS (128 KB, 16B chunks) ----
    {
        const uint4* src = (const uint4*)bpack;
        uint4*       dst = (uint4*)ldsB;
        #pragma unroll
        for (int i = 0; i < (WS_BPACK_BYTES / 16) / 256; ++i)
            dst[tid + i * 256] = src[tid + i * 256];
    }
    if (tid < KC) ldsPrior[tid] = __logf(comm[tid] + EPSF);
    __syncthreads();

    // ---- A addressing: wave owns rows m0..m0+31; lane's rows for the two
    //      M tiles are m0 + (lane&15) and m0 + 16 + (lane&15) (clamped) ----
    const int m0 = blockIdx.x * ROWS_PER_BLOCK + wave * ROWS_PER_WAVE;
    int r0 = m0 + nl;       if (r0 > NP - 1) r0 = NP - 1;
    int r1 = m0 + 16 + nl;  if (r1 > NP - 1) r1 = NP - 1;
    const float* __restrict__ arow0 = counts + (size_t)r0 * OD;
    const float* __restrict__ arow1 = counts + (size_t)r1 * OD;

    v8f c00 = {}, c01 = {}, c02 = {}, c03 = {};   // M tile 0, 4 column tiles
    v8f c10 = {}, c11 = {}, c12 = {}, c13 = {};   // M tile 1

    for (int kt = 0; kt < KT_COUNT; ++kt) {
        if (kt + 1 < KT_COUNT) {                  // near-scope prefetch of next 128B
            __builtin_prefetch(arow0 + (kt + 1) * 32 + 8 * half, 0, 3);
            __builtin_prefetch(arow1 + (kt + 1) * 32 + 8 * half, 0, 3);
        }

        // build A fragments: 16x32 f16, ISA layout (f32 -> f16 exact: 0..9)
        v16h a0, a1;
        const int kbase = kt * 32 + 8 * half;
        #pragma unroll
        for (int v = 0; v < 8; ++v) {
            int k = kbase + (v < 4 ? 2 * v : 16 + 2 * (v - 4));
            float2 x0 = *(const float2*)(arow0 + k);
            float2 x1 = *(const float2*)(arow1 + k);
            a0[2 * v]     = (_Float16)x0.x;
            a0[2 * v + 1] = (_Float16)x0.y;
            a1[2 * v]     = (_Float16)x1.x;
            a1[2 * v + 1] = (_Float16)x1.y;
        }

        // B fragments from LDS, shared by both M tiles; 8 WMMAs per k-step
        const _Float16* bf = ldsB + ((size_t)(kt * CT_COUNT) * 32 + lane) * 16;
        v16h b0 = *(const v16h*)(bf);
        c00 = __builtin_amdgcn_wmma_f32_16x16x32_f16(false, a0, false, b0, (short)0, c00, false, false);
        c10 = __builtin_amdgcn_wmma_f32_16x16x32_f16(false, a1, false, b0, (short)0, c10, false, false);
        v16h b1 = *(const v16h*)(bf + 512);
        c01 = __builtin_amdgcn_wmma_f32_16x16x32_f16(false, a0, false, b1, (short)0, c01, false, false);
        c11 = __builtin_amdgcn_wmma_f32_16x16x32_f16(false, a1, false, b1, (short)0, c11, false, false);
        v16h b2 = *(const v16h*)(bf + 1024);
        c02 = __builtin_amdgcn_wmma_f32_16x16x32_f16(false, a0, false, b2, (short)0, c02, false, false);
        c12 = __builtin_amdgcn_wmma_f32_16x16x32_f16(false, a1, false, b2, (short)0, c12, false, false);
        v16h b3 = *(const v16h*)(bf + 1536);
        c03 = __builtin_amdgcn_wmma_f32_16x16x32_f16(false, a0, false, b3, (short)0, c03, false, false);
        c13 = __builtin_amdgcn_wmma_f32_16x16x32_f16(false, a1, false, b3, (short)0, c13, false, false);
    }

    // all waves done reading B -> safe to reuse its LDS as the row buffer
    __syncthreads();

    // C layout: VGPR r holds M = r + 8*half (+16 for tile 1), N = nl + 16*ct
    float* myrows = ldsRows + wave * (ROWS_PER_WAVE * KC);   // 8 KB per wave
    {
        float lp0 = ldsPrior[nl +  0];
        float lp1 = ldsPrior[nl + 16];
        float lp2 = ldsPrior[nl + 32];
        float lp3 = ldsPrior[nl + 48];
        #pragma unroll
        for (int r = 0; r < 8; ++r) {
            int rl0 = r + 8 * half;          // tile 0 local row
            int rl1 = rl0 + 16;              // tile 1 local row
            myrows[rl0 * KC + nl +  0] = c00[r] + lp0;
            myrows[rl0 * KC + nl + 16] = c01[r] + lp1;
            myrows[rl0 * KC + nl + 32] = c02[r] + lp2;
            myrows[rl0 * KC + nl + 48] = c03[r] + lp3;
            myrows[rl1 * KC + nl +  0] = c10[r] + lp0;
            myrows[rl1 * KC + nl + 16] = c11[r] + lp1;
            myrows[rl1 * KC + nl + 32] = c12[r] + lp2;
            myrows[rl1 * KC + nl + 48] = c13[r] + lp3;
        }
    }
    // intra-wave LDS ops are in-order; each wave reads only its own region

    // per-lane row logsumexp: lane handles local row == lane (all 32 busy)
    float rowval = 0.0f;
    {
        int grow = m0 + lane;
        if (grow < NP) {
            const float* rp = myrows + lane * KC;
            float m = rp[0];
            #pragma unroll 8
            for (int j = 1; j < KC; ++j) m = fmaxf(m, rp[j]);
            float s = 0.0f;
            #pragma unroll 8
            for (int j = 0; j < KC; ++j) s += __expf(rp[j] - m);
            rowval = m + __logf(s);
        }
    }
    // full wave32 butterfly reduction
    #pragma unroll
    for (int off = 16; off >= 1; off >>= 1)
        rowval += __shfl_xor(rowval, off, 32);
    if (lane == 0) ldsWave[wave] = rowval;
    __syncthreads();
    if (tid == 0) {
        float s = 0.0f;
        #pragma unroll
        for (int w = 0; w < WAVES_PER_BLOCK; ++w) s += ldsWave[w];
        partials[blockIdx.x] = s;
    }
}

// ---------------------------------------------------------------------------
// Kernel 3: deterministic fixed-order reduction of block partials.
// ---------------------------------------------------------------------------
__global__ __launch_bounds__(256) void reduce_partials(const float* __restrict__ partials,
                                                       float* __restrict__ out, int n) {
    __shared__ float sm[256];
    int t = threadIdx.x;
    float s = 0.0f;
    for (int i = t; i < n; i += 256) s += partials[i];
    sm[t] = s;
    __syncthreads();
    #pragma unroll
    for (int off = 128; off >= 1; off >>= 1) {
        if (t < off) sm[t] += sm[t + off];
        __syncthreads();
    }
    if (t == 0) out[0] = sm[0];
}

extern "C" void kernel_launch(void* const* d_in, const int* in_sizes, int n_in,
                              void* d_out, int out_size, void* d_ws, size_t ws_size,
                              hipStream_t stream) {
    const float* counts = (const float*)d_in[0];   // (N, O) f32
    const float* otu    = (const float*)d_in[1];   // (K, O) f32
    const float* comm   = (const float*)d_in[2];   // (K,)   f32
    float* out = (float*)d_out;

    _Float16* bpack    = (_Float16*)d_ws;
    float*    partials = (float*)((char*)d_ws + WS_BPACK_BYTES);

    pack_logB<<<BPACK_HALFS / 256, 256, 0, stream>>>(otu, bpack);

    size_t shmem = WS_BPACK_BYTES + 256 + 32;      // B/rows + prior + wave partials
    ll_fused<<<NBLOCKS, 256, shmem, stream>>>(counts, comm, bpack, partials);

    reduce_partials<<<1, 256, 0, stream>>>(partials, out, NBLOCKS);
}
